// SKiT_22960895165058
// MI455X (gfx1250) — compile-verified, matrix operand order checked by reading, not run on hardware
//
#include <hip/hip_runtime.h>
#include <math.h>

// ---------------- model dims ----------------
#define S_   2048
#define VIT_ 768
#define D_   512
#define NH_  8
#define DH_  64
#define FF_  2048
#define GF_  64
#define NC_  14
#define NL_  2

// Master switch for the CDNA5 async global->LDS path
#define TRY_ASYNC 1

#if TRY_ASYNC && defined(__HIP_DEVICE_COMPILE__) && \
    __has_builtin(__builtin_amdgcn_global_load_async_to_lds_b128)
#define USE_ASYNC_LDS 1
#else
#define USE_ASYNC_LDS 0
#endif

typedef unsigned short ushort_t;

// vector-of-4-int type matching the async builtin's pointee type
typedef int v4i __attribute__((__vector_size__(4 * sizeof(int))));

// ---------------- WMMA types ----------------
typedef __attribute__((ext_vector_type(16))) __bf16 bf16x16;
typedef __attribute__((ext_vector_type(8)))  float  f32x8;

__device__ __forceinline__ unsigned short f2bf(float f) {
    unsigned int u = __float_as_uint(f);
    u += 0x7FFFu + ((u >> 16) & 1u);   // round-to-nearest-even
    return (unsigned short)(u >> 16);
}

// pack two floats -> two bf16 in one dword, staying in 32-bit registers
__device__ __forceinline__ unsigned int pk2(float lo, float hi) {
    unsigned int a = __float_as_uint(lo), b = __float_as_uint(hi);
    a += 0x7FFFu + ((a >> 16) & 1u);
    b += 0x7FFFu + ((b >> 16) & 1u);
    return (a >> 16) | (b & 0xFFFF0000u);
}

__device__ __forceinline__ float bf2f(unsigned short u) {
    return __uint_as_float((unsigned int)u << 16);
}
__device__ __forceinline__ void unp(unsigned int u, float& lo, float& hi) {
    lo = __uint_as_float(u << 16);
    hi = __uint_as_float(u & 0xFFFF0000u);
}

// 16-byte global -> LDS copy (async TDM-style path when available)
__device__ __forceinline__ void cp16(const void* gsrc, void* ldst) {
#if USE_ASYNC_LDS
    __builtin_amdgcn_global_load_async_to_lds_b128(
        (__attribute__((address_space(1))) v4i*)(unsigned long long)(size_t)gsrc,
        (__attribute__((address_space(3))) v4i*)(unsigned int)(size_t)ldst, 0, 0);
#else
    *(uint4*)ldst = *(const uint4*)gsrc;
#endif
}
__device__ __forceinline__ void cp_wait() {
#if USE_ASYNC_LDS
#if __has_builtin(__builtin_amdgcn_s_wait_asynccnt)
    __builtin_amdgcn_s_wait_asynccnt(0);
#else
    asm volatile("s_wait_asynccnt 0x0" ::: "memory");
#endif
#endif
}

union FragU { uint4 u[2]; bf16x16 v; };

// Operand fragment for one lane from an LDS row (32-wide K slice):
// lane holds K = kk..kk+7 and 16+kk..16+kk+7 -> two ds_load_b128.
__device__ __forceinline__ bf16x16 ldfrag(const unsigned short* rowbase, int kk) {
    FragU f;
    f.u[0] = *(const uint4*)(rowbase + kk);
    f.u[1] = *(const uint4*)(rowbase + 16 + kk);
    return f.v;
}

__device__ __forceinline__ f32x8 zero8() {
    f32x8 z;
#pragma unroll
    for (int i = 0; i < 8; ++i) z[i] = 0.0f;
    return z;
}

__device__ __forceinline__ f32x8 wmma_bf16(bf16x16 a, bf16x16 b, f32x8 c) {
    return __builtin_amdgcn_wmma_f32_16x16x32_bf16(false, a, false, b, (short)0, c,
                                                   false, false);
}

// =====================================================================
// fp32 -> bf16 bulk convert (weights / input frame), 4 elems per thread
// =====================================================================
__global__ __launch_bounds__(256)
void cvt_bf16_kernel(const float* __restrict__ src, ushort_t* __restrict__ dst,
                     int n4) {
    int i = blockIdx.x * blockDim.x + threadIdx.x;
    if (i >= n4) return;
    float4 f = ((const float4*)src)[i];
    uint2 p; p.x = pk2(f.x, f.y); p.y = pk2(f.z, f.w);
    ((uint2*)dst)[i] = p;
}

// =====================================================================
// GEMM: C[M,N] = act(A[M,K] @ W[N,K]^T + bias + resid). A,W,resid bf16;
// bias fp32; C bf16 or fp32. Tile 128x64, BK=64, 256 thr = 8 waves
// (4m x 2n), 2x2 accumulators, 8 WMMAs per k-step. Double-buffered LDS:
// async copies for tile t+1 overlap WMMA compute on tile t.
// grid = (M/128, ceil(N/64)); M mult of 128, K mult of 64, lda mult of 8.
// =====================================================================
__global__ __launch_bounds__(256)
void gemm_wmma_kernel(const ushort_t* __restrict__ A, int lda,
                      const ushort_t* __restrict__ W, int ldw,
                      const float* __restrict__ bias,
                      const ushort_t* __restrict__ resid, int ldr,
                      void* __restrict__ Cv, int ldc,
                      int N, int K, int relu, int c_bf16) {
    __shared__ unsigned short As[2][128 * 64];   // 2 x 16 KB
    __shared__ unsigned short Bs[2][64 * 64];    // 2 x  8 KB

    const int t    = threadIdx.x;
    const int m0   = blockIdx.x * 128;
    const int n0   = blockIdx.y * 64;
    const int wave = t >> 5;
    const int lane = t & 31;
    const int wm   = wave & 3;
    const int wn   = wave >> 2;
    const int l15  = lane & 15;
    const int kk   = (lane & 16) ? 8 : 0;

    auto stage = [&](int buf, int k0) {
#pragma unroll
        for (int it = 0; it < 4; ++it) {
            int ch = t + it * 256;
            int r = ch >> 3, c8 = (ch & 7) * 8;
            cp16(A + (size_t)(m0 + r) * lda + k0 + c8, &As[buf][r * 64 + c8]);
        }
        // clamp row for n >= N (results discarded in epilogue)
#pragma unroll
        for (int it = 0; it < 2; ++it) {
            int ch = t + it * 256;
            int r = ch >> 3, c8 = (ch & 7) * 8;
            int n = n0 + r;
            int nc = (n < N) ? n : (N - 1);
            cp16(W + (size_t)nc * ldw + k0 + c8, &Bs[buf][r * 64 + c8]);
        }
    };

    f32x8 acc[2][2];
#pragma unroll
    for (int a = 0; a < 2; ++a)
#pragma unroll
        for (int b = 0; b < 2; ++b) acc[a][b] = zero8();

    stage(0, 0);
    cp_wait();
    __syncthreads();

    int cur = 0;
    for (int k0 = 0; k0 < K; k0 += 64) {
        if (k0 + 64 < K) stage(cur ^ 1, k0 + 64);   // prefetch next tile

        bf16x16 af[2][2], bfr[2][2];
#pragma unroll
        for (int kt = 0; kt < 2; ++kt) {
            af[0][kt]  = ldfrag(&As[cur][(wm * 32 + l15) * 64 + kt * 32], kk);
            af[1][kt]  = ldfrag(&As[cur][(wm * 32 + 16 + l15) * 64 + kt * 32], kk);
            bfr[0][kt] = ldfrag(&Bs[cur][(wn * 32 + l15) * 64 + kt * 32], kk);
            bfr[1][kt] = ldfrag(&Bs[cur][(wn * 32 + 16 + l15) * 64 + kt * 32], kk);
        }
#pragma unroll
        for (int a = 0; a < 2; ++a)
#pragma unroll
            for (int b = 0; b < 2; ++b)
#pragma unroll
                for (int kt = 0; kt < 2; ++kt)
                    acc[a][b] = wmma_bf16(af[a][kt], bfr[b][kt], acc[a][b]);

        cp_wait();        // next tile's async copies complete
        __syncthreads();  // and everyone is done reading cur
        cur ^= 1;
    }

    // epilogue: lane n = lane&15, VGPR r -> M = r + 8*(lane>=16)
#pragma unroll
    for (int a = 0; a < 2; ++a)
#pragma unroll
        for (int b = 0; b < 2; ++b) {
            int n = n0 + wn * 32 + b * 16 + l15;
            if (n >= N) continue;
            float bv = bias ? bias[n] : 0.0f;
#pragma unroll
            for (int r = 0; r < 8; ++r) {
                int m = m0 + wm * 32 + a * 16 + r + kk;
                float v = acc[a][b][r] + bv;
                if (resid) v += bf2f(resid[(size_t)m * ldr + n]);
                if (relu)  v = fmaxf(v, 0.0f);
                if (c_bf16) ((ushort_t*)Cv)[(size_t)m * ldc + n] = f2bf(v);
                else        ((float*)Cv)[(size_t)m * ldc + n] = v;
            }
        }
}

// =====================================================================
// Flash attention (bf16 in/out): one block = 64 query rows x one head.
// 128 threads = 4 waves x 16 query rows. Streams K/V in 64-key tiles
// with double-buffered staging. grid = (S/64, NH).
// =====================================================================
__global__ __launch_bounds__(128)
void flash_attn_kernel(const ushort_t* __restrict__ Qp,
                       const ushort_t* __restrict__ Kp,
                       const ushort_t* __restrict__ Vp, int ld,
                       ushort_t* __restrict__ O, int ldo,
                       int Slen, float scale) {
    const int h    = blockIdx.y;
    const int m0   = blockIdx.x * 64;
    const int t    = threadIdx.x;
    const int wave = t >> 5;
    const int lane = t & 31;
    const int l15  = lane & 15;
    const int kk   = (lane & 16) ? 8 : 0;

    const ushort_t* Qh = Qp + h * DH_;
    const ushort_t* Kh = Kp + h * DH_;
    const ushort_t* Vh = Vp + h * DH_;

    __shared__ unsigned short sQ[64 * 64];        //  8 KB
    __shared__ unsigned short sK[2][64 * 64];     // 16 KB
    __shared__ unsigned short sVt[2][64 * 64];    // 16 KB
    __shared__ unsigned short sP[4][16 * 64];     //  8 KB

    auto stageKV = [&](int buf, int kb) {
#pragma unroll
        for (int it = 0; it < 4; ++it) {          // K tile: [key][d], pure copy
            int ch = t + it * 128;
            int r = ch >> 3, c8 = (ch & 7) * 8;
            cp16(Kh + (size_t)(kb + r) * ld + c8, &sK[buf][r * 64 + c8]);
        }
#pragma unroll
        for (int it = 0; it < 8; ++it) {          // V tile transposed: [d][key]
            int ch = t + it * 128;
            int r = ch >> 4, c4 = (ch & 15) * 4;
            uint2 p = *(const uint2*)(Vh + (size_t)(kb + r) * ld + c4);
            sVt[buf][(c4 + 0) * 64 + r] = (unsigned short)p.x;
            sVt[buf][(c4 + 1) * 64 + r] = (unsigned short)(p.x >> 16);
            sVt[buf][(c4 + 2) * 64 + r] = (unsigned short)p.y;
            sVt[buf][(c4 + 3) * 64 + r] = (unsigned short)(p.y >> 16);
        }
    };

    // stage Q once + first K/V tile
#pragma unroll
    for (int it = 0; it < 4; ++it) {
        int ch = t + it * 128;
        int r = ch >> 3, c8 = (ch & 7) * 8;
        cp16(Qh + (size_t)(m0 + r) * ld + c8, &sQ[r * 64 + c8]);
    }
    stageKV(0, 0);
    cp_wait();
    __syncthreads();

    bf16x16 qf[2];
    qf[0] = ldfrag(&sQ[(wave * 16 + l15) * 64], kk);       // d = 0..31
    qf[1] = ldfrag(&sQ[(wave * 16 + l15) * 64 + 32], kk);  // d = 32..63

    float mr[8], lr[8];
#pragma unroll
    for (int r = 0; r < 8; ++r) { mr[r] = -__builtin_inff(); lr[r] = 0.0f; }
    f32x8 o[4];
#pragma unroll
    for (int dt = 0; dt < 4; ++dt) o[dt] = zero8();

    int cur = 0;
    for (int kb = 0; kb < Slen; kb += 64) {
        if (kb + 64 < Slen) stageKV(cur ^ 1, kb + 64);  // prefetch next tile

        // S = Q @ K^T (4 key sub-tiles of 16), then fold in 1/sqrt(dh)
        f32x8 sc[4];
#pragma unroll
        for (int nt = 0; nt < 4; ++nt) {
            f32x8 s = zero8();
            bf16x16 b0 = ldfrag(&sK[cur][(nt * 16 + l15) * 64], kk);
            bf16x16 b1 = ldfrag(&sK[cur][(nt * 16 + l15) * 64 + 32], kk);
            s = wmma_bf16(qf[0], b0, s);
            s = wmma_bf16(qf[1], b1, s);
            sc[nt] = s * scale;
        }

        unsigned short* pw = &sP[wave][0];
#pragma unroll
        for (int r = 0; r < 8; ++r) {
            float tm = fmaxf(fmaxf(sc[0][r], sc[1][r]), fmaxf(sc[2][r], sc[3][r]));
            tm = fmaxf(tm, __shfl_xor(tm, 1, 32));
            tm = fmaxf(tm, __shfl_xor(tm, 2, 32));
            tm = fmaxf(tm, __shfl_xor(tm, 4, 32));
            tm = fmaxf(tm, __shfl_xor(tm, 8, 32));
            float nm    = fmaxf(mr[r], tm);
            float alpha = __expf(mr[r] - nm);
            float rs    = 0.0f;
            int mloc    = r + kk;  // same 0/8 offset as C layout
#pragma unroll
            for (int nt = 0; nt < 4; ++nt) {
                float p = __expf(sc[nt][r] - nm);
                rs += p;
                pw[mloc * 64 + nt * 16 + l15] = f2bf(p);
            }
            rs += __shfl_xor(rs, 1, 32);
            rs += __shfl_xor(rs, 2, 32);
            rs += __shfl_xor(rs, 4, 32);
            rs += __shfl_xor(rs, 8, 32);
            lr[r] = lr[r] * alpha + rs;
            mr[r] = nm;
#pragma unroll
            for (int dt = 0; dt < 4; ++dt) o[dt][r] *= alpha;
        }
        __syncthreads();  // order sP stores before A-layout reloads

        bf16x16 pf0 = ldfrag(&pw[l15 * 64], kk);       // keys 0..31
        bf16x16 pf1 = ldfrag(&pw[l15 * 64 + 32], kk);  // keys 32..63
#pragma unroll
        for (int dt = 0; dt < 4; ++dt) {
            bf16x16 v0 = ldfrag(&sVt[cur][(dt * 16 + l15) * 64], kk);
            bf16x16 v1 = ldfrag(&sVt[cur][(dt * 16 + l15) * 64 + 32], kk);
            o[dt] = wmma_bf16(pf0, v0, o[dt]);
            o[dt] = wmma_bf16(pf1, v1, o[dt]);
        }

        cp_wait();        // next tile's async copies complete
        __syncthreads();  // and everyone is done reading cur
        cur ^= 1;
    }

#pragma unroll
    for (int dt = 0; dt < 4; ++dt)
#pragma unroll
        for (int r = 0; r < 8; ++r) {
            int m = m0 + wave * 16 + r + kk;
            int n = h * DH_ + dt * 16 + l15;
            O[(size_t)m * ldo + n] = f2bf(o[dt][r] / lr[r]);
        }
}

// =====================================================================
// y = LayerNorm(x + h) * s + b ; bf16 in/out, fp32 math. One wave per
// 512-wide row, each lane owns 16 contiguous elems. grid = S/8, 256 thr.
// =====================================================================
__global__ __launch_bounds__(256)
void ln_res_kernel(const ushort_t* __restrict__ x, const ushort_t* __restrict__ hh,
                   const float* __restrict__ s, const float* __restrict__ b,
                   ushort_t* __restrict__ y) {
    const int wave = threadIdx.x >> 5, lane = threadIdx.x & 31;
    const int row  = blockIdx.x * 8 + wave;
    const int c0   = lane * 16;
    const uint4* xv = (const uint4*)(x  + (size_t)row * D_ + c0);
    const uint4* hv = (const uint4*)(hh + (size_t)row * D_ + c0);
    uint4 xa = xv[0], xb = xv[1], ha = hv[0], hb = hv[1];
    unsigned int xs[8] = {xa.x, xa.y, xa.z, xa.w, xb.x, xb.y, xb.z, xb.w};
    unsigned int hs[8] = {ha.x, ha.y, ha.z, ha.w, hb.x, hb.y, hb.z, hb.w};
    float v[16], sum = 0.0f;
#pragma unroll
    for (int j = 0; j < 8; ++j) {
        float xl, xh2, hl, hh2;
        unp(xs[j], xl, xh2);
        unp(hs[j], hl, hh2);
        v[2*j]   = xl + hl;
        v[2*j+1] = xh2 + hh2;
        sum += v[2*j] + v[2*j+1];
    }
    for (int m = 16; m >= 1; m >>= 1) sum += __shfl_xor(sum, m, 32);
    float mean = sum * (1.0f / (float)D_);
    float var = 0.0f;
#pragma unroll
    for (int j = 0; j < 16; ++j) { float d = v[j] - mean; var += d * d; }
    for (int m = 16; m >= 1; m >>= 1) var += __shfl_xor(var, m, 32);
    float rstd = rsqrtf(var * (1.0f / (float)D_) + 1e-5f);
    float r16[16];
#pragma unroll
    for (int j = 0; j < 4; ++j) {
        float4 sc = *(const float4*)(s + c0 + j * 4);
        float4 bb = *(const float4*)(b + c0 + j * 4);
        r16[j*4+0] = (v[j*4+0] - mean) * rstd * sc.x + bb.x;
        r16[j*4+1] = (v[j*4+1] - mean) * rstd * sc.y + bb.y;
        r16[j*4+2] = (v[j*4+2] - mean) * rstd * sc.z + bb.z;
        r16[j*4+3] = (v[j*4+3] - mean) * rstd * sc.w + bb.w;
    }
    uint4 oa, ob;
    oa.x = pk2(r16[0],  r16[1]);  oa.y = pk2(r16[2],  r16[3]);
    oa.z = pk2(r16[4],  r16[5]);  oa.w = pk2(r16[6],  r16[7]);
    ob.x = pk2(r16[8],  r16[9]);  ob.y = pk2(r16[10], r16[11]);
    ob.z = pk2(r16[12], r16[13]); ob.w = pk2(r16[14], r16[15]);
    uint4* yv = (uint4*)(y + (size_t)row * D_ + c0);
    yv[0] = oa; yv[1] = ob;
}

// =====================================================================
// Column-wise cumulative max along S (bf16 in/out). One block per column.
// =====================================================================
__global__ __launch_bounds__(256)
void cummax_kernel(const ushort_t* __restrict__ src, ushort_t* __restrict__ dst,
                   int cols) {
    const int col = blockIdx.x, t = threadIdx.x;
    float v[8], run = -__builtin_inff();
    const int base = t * 8;
#pragma unroll
    for (int i = 0; i < 8; ++i) {
        run = fmaxf(run, bf2f(src[(size_t)(base + i) * cols + col]));
        v[i] = run;
    }
    __shared__ float part[256];
    part[t] = run;
    __syncthreads();
    for (int off = 1; off < 256; off <<= 1) {
        float a = part[t];
        float b = (t >= off) ? part[t - off] : -__builtin_inff();
        __syncthreads();
        part[t] = fmaxf(a, b);
        __syncthreads();
    }
    float excl = (t == 0) ? -__builtin_inff() : part[t - 1];
#pragma unroll
    for (int i = 0; i < 8; ++i)
        dst[(size_t)(base + i) * cols + col] = f2bf(fmaxf(excl, v[i]));
}

// row softmax, in place, one thread per row (fp32, N = 14)
__global__ __launch_bounds__(256)
void softmax_rows_kernel(float* __restrict__ x, int M, int N) {
    int r = blockIdx.x * blockDim.x + threadIdx.x;
    if (r >= M) return;
    float* row = x + (size_t)r * N;
    float mx = -__builtin_inff();
    for (int n = 0; n < N; ++n) mx = fmaxf(mx, row[n]);
    float ssum = 0.0f;
    for (int n = 0; n < N; ++n) { float e = __expf(row[n] - mx); row[n] = e; ssum += e; }
    float inv = 1.0f / ssum;
    for (int n = 0; n < N; ++n) row[n] *= inv;
}

// =====================================================================
// Host orchestration
// =====================================================================
extern "C" void kernel_launch(void* const* d_in, const int* in_sizes, int n_in,
                              void* d_out, int out_size, void* d_ws, size_t ws_size,
                              hipStream_t stream) {
    (void)in_sizes; (void)n_in; (void)out_size; (void)ws_size;
    const float* frame = (const float*)d_in[0];
    const float* W0 = (const float*)d_in[2];   const float* b0 = (const float*)d_in[3];
    const float* W1 = (const float*)d_in[4];   const float* b1 = (const float*)d_in[5];
    const float* W2 = (const float*)d_in[6];   const float* b2 = (const float*)d_in[7];
    const float* W3 = (const float*)d_in[8];   const float* b3 = (const float*)d_in[9];
    const float* W4 = (const float*)d_in[10];  const float* b4 = (const float*)d_in[11];
    const float* W5 = (const float*)d_in[12];  const float* b5 = (const float*)d_in[13];
    const float* enc_qkv_w = (const float*)d_in[14];
    const float* enc_qkv_b = (const float*)d_in[15];
    const float* enc_out_w = (const float*)d_in[16];
    const float* enc_out_b = (const float*)d_in[17];
    const float* sa_qkv_w  = (const float*)d_in[18];
    const float* sa_qkv_b  = (const float*)d_in[19];
    const float* sa_out_w  = (const float*)d_in[20];
    const float* sa_out_b  = (const float*)d_in[21];
    const float* ca_qkv_w  = (const float*)d_in[22];
    const float* ca_qkv_b  = (const float*)d_in[23];
    const float* ca_out_w  = (const float*)d_in[24];
    const float* ca_out_b  = (const float*)d_in[25];
    const float* enc_ff1_w = (const float*)d_in[26];
    const float* enc_ff1_b = (const float*)d_in[27];
    const float* enc_ff2_w = (const float*)d_in[28];
    const float* enc_ff2_b = (const float*)d_in[29];
    const float* dec_ff1_w = (const float*)d_in[30];
    const float* dec_ff1_b = (const float*)d_in[31];
    const float* dec_ff2_w = (const float*)d_in[32];
    const float* dec_ff2_b = (const float*)d_in[33];
    const float* enc_ln1_s = (const float*)d_in[34];
    const float* enc_ln1_b = (const float*)d_in[35];
    const float* enc_ln2_s = (const float*)d_in[36];
    const float* enc_ln2_b = (const float*)d_in[37];
    const float* dec_ln1_s = (const float*)d_in[38];
    const float* dec_ln1_b = (const float*)d_in[39];
    const float* dec_ln2_s = (const float*)d_in[40];
    const float* dec_ln2_b = (const float*)d_in[41];
    const float* dec_ln3_s = (const float*)d_in[42];
    const float* dec_ln3_b = (const float*)d_in[43];

    // ---- workspace carve (bf16 elements) ----
    ushort_t* w = (ushort_t*)d_ws;
    auto take = [&](size_t n) { ushort_t* p = w; w += n; return p; };
    ushort_t* cFrame  = take((size_t)S_ * VIT_);
    ushort_t* cW0 = take((size_t)D_ * VIT_);
    ushort_t* cW1 = take((size_t)NC_ * D_);
    ushort_t* cW2 = take((size_t)NC_ * D_);
    ushort_t* cW3 = take((size_t)GF_ * D_);
    ushort_t* cW4 = take((size_t)D_ * GF_);
    ushort_t* cW5 = take((size_t)D_ * D_);
    ushort_t* cEncQkv = take((size_t)NL_ * 3 * D_ * D_);
    ushort_t* cEncOut = take((size_t)NL_ * D_ * D_);
    ushort_t* cSaQkv  = take((size_t)NL_ * 3 * D_ * D_);
    ushort_t* cSaOut  = take((size_t)NL_ * D_ * D_);
    ushort_t* cCaQkv  = take((size_t)NL_ * 3 * D_ * D_);
    ushort_t* cCaOut  = take((size_t)NL_ * D_ * D_);
    ushort_t* cEncF1  = take((size_t)NL_ * FF_ * D_);
    ushort_t* cEncF2  = take((size_t)NL_ * D_ * FF_);
    ushort_t* cDecF1  = take((size_t)NL_ * FF_ * D_);
    ushort_t* cDecF2  = take((size_t)NL_ * D_ * FF_);
    ushort_t* bx   = take((size_t)S_ * D_);
    ushort_t* bmem = take((size_t)S_ * D_);
    ushort_t* btgt = take((size_t)S_ * D_);
    ushort_t* bqkv = take((size_t)S_ * 3 * D_);
    ushort_t* batt = take((size_t)S_ * D_);
    ushort_t* bh   = take((size_t)S_ * D_);
    ushort_t* bff  = take((size_t)S_ * FF_);
    ushort_t* bloc = take((size_t)S_ * GF_);
    ushort_t* bseq = take((size_t)S_ * GF_);
    ushort_t* bo1  = take((size_t)S_ * D_);
    ushort_t* bo2  = take((size_t)S_ * D_);

    auto cvt = [&](const float* src, ushort_t* dst, size_t nelem) {
        int n4 = (int)(nelem / 4);
        cvt_bf16_kernel<<<(n4 + 255) / 256, 256, 0, stream>>>(src, dst, n4);
    };
    cvt(frame, cFrame, (size_t)S_ * VIT_);
    cvt(W0, cW0, (size_t)D_ * VIT_);   cvt(W1, cW1, (size_t)NC_ * D_);
    cvt(W2, cW2, (size_t)NC_ * D_);    cvt(W3, cW3, (size_t)GF_ * D_);
    cvt(W4, cW4, (size_t)D_ * GF_);    cvt(W5, cW5, (size_t)D_ * D_);
    cvt(enc_qkv_w, cEncQkv, (size_t)NL_ * 3 * D_ * D_);
    cvt(enc_out_w, cEncOut, (size_t)NL_ * D_ * D_);
    cvt(sa_qkv_w,  cSaQkv,  (size_t)NL_ * 3 * D_ * D_);
    cvt(sa_out_w,  cSaOut,  (size_t)NL_ * D_ * D_);
    cvt(ca_qkv_w,  cCaQkv,  (size_t)NL_ * 3 * D_ * D_);
    cvt(ca_out_w,  cCaOut,  (size_t)NL_ * D_ * D_);
    cvt(enc_ff1_w, cEncF1,  (size_t)NL_ * FF_ * D_);
    cvt(enc_ff2_w, cEncF2,  (size_t)NL_ * D_ * FF_);
    cvt(dec_ff1_w, cDecF1,  (size_t)NL_ * FF_ * D_);
    cvt(dec_ff2_w, cDecF2,  (size_t)NL_ * D_ * FF_);

    auto gemm = [&](const ushort_t* A, int lda, const ushort_t* Wt, int ldw,
                    const float* bias, const ushort_t* resid, int ldr,
                    void* C, int ldc, int N, int K, int relu, int cbf) {
        dim3 g(S_ / 128, (N + 63) / 64);
        gemm_wmma_kernel<<<g, 256, 0, stream>>>(A, lda, Wt, ldw, bias, resid, ldr,
                                                C, ldc, N, K, relu, cbf);
    };
    auto flash = [&](const ushort_t* q, const ushort_t* k, const ushort_t* v,
                     int ld, ushort_t* o) {
        flash_attn_kernel<<<dim3(S_ / 64, NH_), 128, 0, stream>>>(q, k, v, ld, o, D_,
                                                                  S_, 0.125f);
    };
    auto ln = [&](const ushort_t* x, const ushort_t* h, const float* s,
                  const float* b, ushort_t* y) {
        ln_res_kernel<<<S_ / 8, 256, 0, stream>>>(x, h, s, b, y);
    };

    // x = frame @ W0^T + b0
    gemm(cFrame, VIT_, cW0, VIT_, b0, nullptr, 0, bx, D_, D_, VIT_, 0, 1);

    // ---------------- encoder ----------------
    const ushort_t* cur = bx;
    for (int i = 0; i < NL_; ++i) {
        gemm(cur, D_, cEncQkv + (size_t)i * 3 * D_ * D_, D_,
             enc_qkv_b + (size_t)i * 3 * D_, nullptr, 0, bqkv, 3 * D_, 3 * D_, D_, 0, 1);
        flash(bqkv, bqkv + D_, bqkv + 2 * D_, 3 * D_, batt);
        gemm(batt, D_, cEncOut + (size_t)i * D_ * D_, D_,
             enc_out_b + (size_t)i * D_, nullptr, 0, bh, D_, D_, D_, 0, 1);
        ln(cur, bh, enc_ln1_s + (size_t)i * D_, enc_ln1_b + (size_t)i * D_, bmem);
        gemm(bmem, D_, cEncF1 + (size_t)i * FF_ * D_, D_,
             enc_ff1_b + (size_t)i * FF_, nullptr, 0, bff, FF_, FF_, D_, 1, 1);
        gemm(bff, FF_, cEncF2 + (size_t)i * D_ * FF_, FF_,
             enc_ff2_b + (size_t)i * D_, nullptr, 0, bh, D_, D_, FF_, 0, 1);
        ln(bmem, bh, enc_ln2_s + (size_t)i * D_, enc_ln2_b + (size_t)i * D_, bmem);
        cur = bmem;
    }

    // ---------------- decoder ----------------
    const ushort_t* ct = bx;
    for (int i = 0; i < NL_; ++i) {
        gemm(ct, D_, cSaQkv + (size_t)i * 3 * D_ * D_, D_,
             sa_qkv_b + (size_t)i * 3 * D_, nullptr, 0, bqkv, 3 * D_, 3 * D_, D_, 0, 1);
        flash(bqkv, bqkv + D_, bqkv + 2 * D_, 3 * D_, batt);
        gemm(batt, D_, cSaOut + (size_t)i * D_ * D_, D_,
             sa_out_b + (size_t)i * D_, nullptr, 0, bh, D_, D_, D_, 0, 1);
        ln(ct, bh, dec_ln1_s + (size_t)i * D_, dec_ln1_b + (size_t)i * D_, btgt);
        gemm(btgt, D_, cCaQkv + (size_t)i * 3 * D_ * D_, D_,
             ca_qkv_b + (size_t)i * 3 * D_, nullptr, 0, bqkv, 3 * D_, D_, D_, 0, 1);
        gemm(bmem, D_, cCaQkv + (size_t)i * 3 * D_ * D_ + (size_t)D_ * D_, D_,
             ca_qkv_b + (size_t)i * 3 * D_ + D_, nullptr, 0, bqkv + D_, 3 * D_,
             2 * D_, D_, 0, 1);
        flash(bqkv, bqkv + D_, bqkv + 2 * D_, 3 * D_, batt);
        gemm(batt, D_, cCaOut + (size_t)i * D_ * D_, D_,
             ca_out_b + (size_t)i * D_, nullptr, 0, bh, D_, D_, D_, 0, 1);
        ln(btgt, bh, dec_ln2_s + (size_t)i * D_, dec_ln2_b + (size_t)i * D_, btgt);
        gemm(btgt, D_, cDecF1 + (size_t)i * FF_ * D_, D_,
             dec_ff1_b + (size_t)i * FF_, nullptr, 0, bff, FF_, FF_, D_, 1, 1);
        gemm(bff, FF_, cDecF2 + (size_t)i * D_ * FF_, FF_,
             dec_ff2_b + (size_t)i * D_, nullptr, 0, bh, D_, D_, FF_, 0, 1);
        ln(btgt, bh, dec_ln3_s + (size_t)i * D_, dec_ln3_b + (size_t)i * D_, btgt);
        ct = btgt;
    }

    // ---------------- head ----------------
    gemm(btgt, D_, cW3, D_, b3, nullptr, 0, bloc, GF_, GF_, D_, 0, 1);
    cummax_kernel<<<GF_, 256, 0, stream>>>(bloc, bseq, GF_);
    gemm(bseq, GF_, cW4, GF_, b4, btgt, D_, bo1, D_, D_, GF_, 0, 1);
    gemm(bo1, D_, cW5, D_, b5, bo1, D_, bo2, D_, D_, D_, 0, 1);

    float* out = (float*)d_out;
    gemm(bo2, D_, cW1, D_, b1, nullptr, 0, out, NC_, NC_, D_, 0, 0);
    softmax_rows_kernel<<<(S_ + 255) / 256, 256, 0, stream>>>(out, S_, NC_);
    gemm(btgt, D_, cW2, D_, b2, nullptr, 0, out + (size_t)S_ * NC_, NC_, NC_, D_, 0, 0);
}
